// VPEncoderByDeForAttDecoder_84061099917455
// MI455X (gfx1250) — matrix operand order repaired and checked by reading, not compiled, not run on hardware
//
#include <hip/hip_runtime.h>
#include <hip/hip_bf16.h>
#include <math.h>

// ---------------- constants ----------------
#define EMBED   256
#define NH      8
#define HD      32          // EMBED / NH
#define NL      4
#define NP      4
#define NLAYERS 3
#define BS      8
#define NQ      1024
#define NV      13294       // 10000 + 2500 + 625 + 169

typedef __bf16 bf16_t;
typedef bf16_t bf16x8  __attribute__((ext_vector_type(8)));
typedef bf16_t bf16x16 __attribute__((ext_vector_type(16)));
typedef float  f32x8   __attribute__((ext_vector_type(8)));
typedef unsigned int u32x4 __attribute__((ext_vector_type(4)));
typedef int          i32x4 __attribute__((ext_vector_type(4)));
typedef int          i32x8 __attribute__((ext_vector_type(8)));

__device__ __constant__ int c_H[NL]    = {100, 50, 25, 13};
__device__ __constant__ int c_base[NL] = {0, 10000, 12500, 13125};

// ---------------- small prep kernels ----------------

// valid ratios: vr[(b*NL+l)*2+0]=vw, +1=vh  (masks are bool -> 1 byte)
__global__ void vr_kernel(const unsigned char* m0, const unsigned char* m1,
                          const unsigned char* m2, const unsigned char* m3,
                          float* __restrict__ vr) {
  int t = threadIdx.x;
  if (t >= BS * NL) return;
  int b = t >> 2, l = t & 3;
  const unsigned char* mp = (l == 0) ? m0 : (l == 1) ? m1 : (l == 2) ? m2 : m3;
  int H = c_H[l], W = c_H[l];
  int sh = 0, sw = 0;
  for (int y = 0; y < H; ++y) sh += (mp[(b * H + y) * W + 0] == 0);
  for (int x = 0; x < W; ++x) sw += (mp[(b * H + 0) * W + x] == 0);
  vr[t * 2 + 0] = (float)sw / (float)W;
  vr[t * 2 + 1] = (float)sh / (float)H;
}

// feats (bs,256,H,W) f32 -> featflat bf16 (bs, NV, 256) rows [base..base+H*W)
__global__ void flatten_kernel(const float* __restrict__ src, bf16_t* __restrict__ dst,
                               int HW, int base) {
  int idx = blockIdx.x * blockDim.x + threadIdx.x;
  int total = BS * HW * EMBED;
  if (idx >= total) return;
  int c = idx % EMBED;
  int r = idx / EMBED;
  int n = r % HW;
  int b = r / HW;
  float v = src[((size_t)b * EMBED + c) * HW + n];
  dst[((size_t)b * NV + base + n) * EMBED + c] = (bf16_t)v;
}

// ref[(b,q,l),2] = (rx*vr[b,l,0], ry*vr[b,l,1])
__global__ void ref_kernel(const float* __restrict__ rp, const float* __restrict__ vr,
                           float* __restrict__ ref) {
  int idx = blockIdx.x * blockDim.x + threadIdx.x;
  if (idx >= BS * NQ * NL) return;
  int l = idx % NL;
  int bq = idx / NL;
  int b = bq / NQ;
  float rx = rp[bq * 2 + 0] / (vr[(b * NL + 0) * 2 + 0] * 100.0f * 8.0f);
  float ry = rp[bq * 2 + 1] / (vr[(b * NL + 0) * 2 + 1] * 100.0f * 8.0f);
  ref[idx * 2 + 0] = rx * vr[(b * NL + l) * 2 + 0];
  ref[idx * 2 + 1] = ry * vr[(b * NL + l) * 2 + 1];
}

// sine embedding of ref level-0 -> qpos (bs,nq,256)
__global__ void qpos_kernel(const float* __restrict__ ref, float* __restrict__ qpos) {
  int idx = blockIdx.x * blockDim.x + threadIdx.x;
  if (idx >= BS * NQ * EMBED) return;
  int j = idx % EMBED;
  int bq = idx / EMBED;
  float x0 = ref[(bq * NL + 0) * 2 + 0];
  float y0 = ref[(bq * NL + 0) * 2 + 1];
  int jj = j & 127;
  float pos = (j < 128) ? y0 : x0;                    // concat(inter(py), inter(px))
  float expo = (2.0f * (float)(jj >> 1)) / 128.0f;
  float dimt = __powf(10000.0f, expo);
  float val = pos * 6.283185307179586f / dimt;
  qpos[idx] = (jj & 1) ? __cosf(val) : __sinf(val);
}

__global__ void init_query_kernel(const float* __restrict__ cls, float* __restrict__ q, int n) {
  int idx = blockIdx.x * blockDim.x + threadIdx.x;
  if (idx < n) q[idx] = cls[idx & (EMBED - 1)];
}

// weight (K=256 x N) f32 -> transposed bf16 (N x 256); coalesced along k on the write side
__global__ void wconv_t_kernel(const float* __restrict__ in, bf16_t* __restrict__ out, int N) {
  int idx = blockIdx.x * blockDim.x + threadIdx.x;
  if (idx >= N * EMBED) return;
  int k = idx % EMBED;
  int n = idx / EMBED;
  out[idx] = (bf16_t)in[(size_t)k * N + n];
}

// q = query + qpos, emitted as bf16 (GEMM A operand)
__global__ void add_to_bf16_kernel(const float* __restrict__ a, const float* __restrict__ b,
                                   bf16_t* __restrict__ out, int n) {
  int idx = blockIdx.x * blockDim.x + threadIdx.x;
  if (idx < n) out[idx] = (bf16_t)(a[idx] + b[idx]);
}

// query = op_out + query (residual); also emit bf16 copy for the FFN GEMM
__global__ void residual_kernel(const float* __restrict__ op_out, float* __restrict__ query,
                                bf16_t* __restrict__ qbf, int n) {
  int idx = blockIdx.x * blockDim.x + threadIdx.x;
  if (idx < n) {
    float v = op_out[idx] + query[idx];
    query[idx] = v;
    qbf[idx] = (bf16_t)v;
  }
}

// ---------------- WMMA GEMM: C(MxN) = A(Mx256) * W(256xN) + bias ----------------
// A bf16 row-major (Mx256); Wt bf16 TRANSPOSED (N x 256). 256 threads = 8 waves.
// Block tile: 128 rows x 64 cols; each wave: 16 rows x 64 cols via 4 accumulators,
// reusing each A fragment across 4 WMMAs. Weight tile (64x256 = 32KB) staged in LDS
// by the Tensor Data Mover (TDM) when available, else a vectorized cooperative copy.
__global__ void __launch_bounds__(256)
gemm_bf16_wmma(const bf16_t* __restrict__ A, const bf16_t* __restrict__ Wt,
               const float* __restrict__ bias, float* __restrict__ C,
               int M, int N) {
  constexpr int K = 256;
  __shared__ bf16_t ldsW[64 * K];                 // [n64][k] : per-column contiguous K
  const int n0 = blockIdx.x * 64;
  const int t  = threadIdx.x;

#if defined(__gfx1250__) && __has_builtin(__builtin_amdgcn_tensor_load_to_lds) && \
    __has_builtin(__builtin_amdgcn_s_wait_tensorcnt)
  if (t < 32) {   // wave 0 drives the TDM; descriptor per CDNA5 ISA D# groups 0/1
    unsigned long long ga = (unsigned long long)(const void*)(Wt + (size_t)n0 * K);
    unsigned lds_byte = (unsigned)(unsigned long long)(const void*)&ldsW[0]; // addr[31:0] = LDS offset
    u32x4 g0;
    g0[0] = 1u;                                        // count=1, user descriptor
    g0[1] = lds_byte;                                  // lds_addr (bytes)
    g0[2] = (unsigned)(ga & 0xFFFFFFFFu);              // global_addr[31:0]
    g0[3] = (unsigned)((ga >> 32) & 0x01FFFFFFu)       // global_addr[56:32]
            | (2u << 30);                              // type = 2 (image)
    i32x8 g1;
    g1[0] = (int)(1u << 16);                           // data_size = 1 (2 bytes)
    g1[1] = (int)((unsigned)(K & 0xFFFF) << 16);       // tensor_dim0[15:0] = 256
    g1[2] = (int)((unsigned)(N & 0xFFFF) << 16);       // tensor_dim1[15:0] = N rows
    g1[3] = (int)((unsigned)(K & 0xFFFF) << 16);       // tile_dim0 = 256
    g1[4] = (int)64;                                   // tile_dim1 = 64 rows, tile_dim2 = 0
    g1[5] = (int)K;                                    // tensor_dim0_stride = 256
    g1[6] = 0;                                         // stride hi / dim1_stride
    g1[7] = 0;
    i32x4 g2 = {0, 0, 0, 0}, g3 = {0, 0, 0, 0};        // 2-D tensor: groups 2/3 unused
#if __clang_major__ >= 23
    i32x8 g4 = {0, 0, 0, 0, 0, 0, 0, 0};
    __builtin_amdgcn_tensor_load_to_lds(g0, g1, g2, g3, g4, 0);
#else
    __builtin_amdgcn_tensor_load_to_lds(g0, g1, g2, g3, 0);
#endif
    __builtin_amdgcn_s_wait_tensorcnt(0);
  }
#else
  {  // fallback: cooperative vectorized copy of the 64x256 bf16 tile (32KB)
    const bf16_t* src = Wt + (size_t)n0 * K;
#pragma unroll
    for (int i = 0; i < 8; ++i) {
      int e = (t + i * 256) * 8;
      *(bf16x8*)(ldsW + e) = *(const bf16x8*)(src + e);
    }
  }
#endif
  __syncthreads();

  const int wv   = t >> 5;
  const int lane = t & 31;
  const int m0   = blockIdx.y * 128 + wv * 16;
  if (m0 >= M) return;                            // wave-uniform (M % 16 == 0)

  const int mrow = m0 + (lane & 15);
  const int kA   = (lane < 16) ? 0 : 8;           // ISA 16-bit A fragment layout
  const int kB   = (lane < 16) ? 0 : 8;           // ISA 16-bit B fragment layout
  const int nn   = lane & 15;
  const bf16_t* arow = A + (size_t)mrow * K + kA;
  const bf16_t* bbase = ldsW + nn * K + kB;       // + j*16*K selects column tile j

  f32x8 acc0 = {}, acc1 = {}, acc2 = {}, acc3 = {};
#pragma unroll
  for (int kk = 0; kk < 8; ++kk) {                // K = 8 * 32
    bf16x8 a0 = *(const bf16x8*)(arow + kk * 32);        // K = kb .. kb+7
    bf16x8 a1 = *(const bf16x8*)(arow + kk * 32 + 16);   // K = kb+16 .. kb+23
    bf16x16 af = __builtin_shufflevector(a0, a1, 0,1,2,3,4,5,6,7,8,9,10,11,12,13,14,15);
#pragma unroll
    for (int j = 0; j < 4; ++j) {
      const bf16_t* bj = bbase + j * 16 * K + kk * 32;
      bf16x8 b0 = *(const bf16x8*)(bj);                  // K = kb .. kb+7
      bf16x8 b1 = *(const bf16x8*)(bj + 16);             // K = kb+16 .. kb+23
      bf16x16 bfv = __builtin_shufflevector(b0, b1, 0,1,2,3,4,5,6,7,8,9,10,11,12,13,14,15);
      f32x8 acc = (j == 0) ? acc0 : (j == 1) ? acc1 : (j == 2) ? acc2 : acc3;
      acc = __builtin_amdgcn_wmma_f32_16x16x32_bf16(false, af, false, bfv,
                                                    (short)0, acc, false, false);
      if (j == 0) acc0 = acc; else if (j == 1) acc1 = acc;
      else if (j == 2) acc2 = acc; else acc3 = acc;
    }
  }

  const int mbase = m0 + ((lane < 16) ? 0 : 8);   // ISA f32 C/D layout: VGPR r -> M = r (+8)
#pragma unroll
  for (int j = 0; j < 4; ++j) {
    const f32x8 acc = (j == 0) ? acc0 : (j == 1) ? acc1 : (j == 2) ? acc2 : acc3;
    const int col = n0 + j * 16 + nn;
    const float bsc = bias[col];
#pragma unroll
    for (int r = 0; r < 8; ++r)
      C[(size_t)(mbase + r) * N + col] = acc[r] + bsc;
  }
}

// ---------------- softmax over NL*NP=16 per (b,q,h) ----------------
__global__ void softmax16_kernel(float* __restrict__ aw) {
  int idx = blockIdx.x * blockDim.x + threadIdx.x;
  if (idx >= BS * NQ * NH) return;
  float* p = aw + (size_t)idx * 16;
  float mx = p[0];
#pragma unroll
  for (int i = 1; i < 16; ++i) mx = fmaxf(mx, p[i]);
  float s = 0.0f;
  float e[16];
#pragma unroll
  for (int i = 0; i < 16; ++i) { e[i] = __expf(p[i] - mx); s += e[i]; }
  float inv = 1.0f / s;
#pragma unroll
  for (int i = 0; i < 16; ++i) p[i] = e[i] * inv;
}

// ---------------- MSDA sampling: one wave per (b,q,h), lane = channel ----------------
__global__ void __launch_bounds__(256)
msda_kernel(const float* __restrict__ v,     // (bs, NV, NH, HD)
            const float* __restrict__ ref,   // (bs, nq, NL, 2)
            const float* __restrict__ so,    // (bs, nq, 256)
            const float* __restrict__ aw,    // (bs, nq, 128) softmaxed
            bf16_t* __restrict__ outb) {     // (bs, nq, 256) bf16
  int gid  = blockIdx.x * blockDim.x + threadIdx.x;
  int wave = gid >> 5;
  int lane = gid & 31;
  if (wave >= BS * NQ * NH) return;
  int h  = wave & (NH - 1);
  int bq = wave >> 3;
  int b  = bq / NQ;

  const float* vb = v + (size_t)b * NV * EMBED + h * HD + lane; // + n*EMBED walks rows
  float acc = 0.0f;
#pragma unroll
  for (int l = 0; l < NL; ++l) {
    const int Wl = c_H[l], Hl = c_H[l], base = c_base[l];
    const float rx = ref[((size_t)bq * NL + l) * 2 + 0];
    const float ry = ref[((size_t)bq * NL + l) * 2 + 1];
#pragma unroll
    for (int p = 0; p < NP; ++p) {
      const int si = (int)((size_t)bq * EMBED) + ((h * NL + l) * NP + p) * 2;
      const float lx = rx + so[si + 0] / (float)Wl;
      const float ly = ry + so[si + 1] / (float)Hl;
      const float w  = aw[(size_t)bq * 128 + h * 16 + l * NP + p];
      const float x  = lx * Wl - 0.5f;
      const float y  = ly * Hl - 0.5f;
      const float x0f = floorf(x), y0f = floorf(y);
      const int x0 = (int)x0f, y0 = (int)y0f;
      const float wx1 = x - x0f, wy1 = y - y0f;
      const float wx0 = 1.0f - wx1, wy0 = 1.0f - wy1;
      float s = 0.0f;
#pragma unroll
      for (int cy = 0; cy < 2; ++cy) {
#pragma unroll
        for (int cx = 0; cx < 2; ++cx) {
          const int xi = x0 + cx, yi = y0 + cy;
          const float cw = (cx ? wx1 : wx0) * (cy ? wy1 : wy0);
          float val = 0.0f;
          if (xi >= 0 && xi < Wl && yi >= 0 && yi < Hl)
            val = vb[(size_t)(base + yi * Wl + xi) * EMBED];
          s += cw * val;
        }
      }
      acc += w * s;
    }
  }
  outb[(size_t)bq * EMBED + h * HD + lane] = (bf16_t)acc;
}

// ---------------- host orchestration ----------------
extern "C" void kernel_launch(void* const* d_in, const int* in_sizes, int n_in,
                              void* d_out, int out_size, void* d_ws, size_t ws_size,
                              hipStream_t stream) {
  (void)in_sizes; (void)n_in; (void)out_size; (void)ws_size;
  const float* feats[4] = {(const float*)d_in[0], (const float*)d_in[3],
                           (const float*)d_in[6], (const float*)d_in[9]};
  const unsigned char* masks[4] = {(const unsigned char*)d_in[2], (const unsigned char*)d_in[5],
                                   (const unsigned char*)d_in[8], (const unsigned char*)d_in[11]};
  const float* ref_points = (const float*)d_in[12];
  const float* cls_embed  = (const float*)d_in[13];
  const float* so_w = (const float*)d_in[15]; const float* so_b = (const float*)d_in[16];
  const float* aw_w = (const float*)d_in[17]; const float* aw_b = (const float*)d_in[18];
  const float* vp_w = (const float*)d_in[19]; const float* vp_b = (const float*)d_in[20];
  const float* op_w = (const float*)d_in[21]; const float* op_b = (const float*)d_in[22];
  const float* ffn_w = (const float*)d_in[23]; const float* ffn_b = (const float*)d_in[24];

  // workspace carve (256B aligned)
  char* ws = (char*)d_ws; size_t off = 0;
  auto carve = [&](size_t bytes) -> void* {
    void* p = ws + off; off = (off + bytes + 255) & ~(size_t)255; return p;
  };
  const int MQ = BS * NQ;                 // 8192
  const int MV = BS * NV;                 // 106352
  bf16_t* featflat = (bf16_t*)carve((size_t)MV * EMBED * sizeof(bf16_t));
  float*  v_buf    = (float*) carve((size_t)MV * EMBED * sizeof(float));
  float*  query    = (float*) carve((size_t)MQ * EMBED * sizeof(float));
  float*  qpos     = (float*) carve((size_t)MQ * EMBED * sizeof(float));
  bf16_t* q_bf     = (bf16_t*)carve((size_t)MQ * EMBED * sizeof(bf16_t));
  float*  so_buf   = (float*) carve((size_t)MQ * EMBED * sizeof(float));
  float*  aw_buf   = (float*) carve((size_t)MQ * 128 * sizeof(float));
  bf16_t* msda_bf  = (bf16_t*)carve((size_t)MQ * EMBED * sizeof(bf16_t));
  float*  op_out   = (float*) carve((size_t)MQ * EMBED * sizeof(float));
  bf16_t* qn_bf    = (bf16_t*)carve((size_t)MQ * EMBED * sizeof(bf16_t));
  bf16_t* wt_bf    = (bf16_t*)carve((size_t)EMBED * EMBED * sizeof(bf16_t)); // W^T (N x K)
  float*  ref_buf  = (float*) carve((size_t)MQ * NL * 2 * sizeof(float));
  float*  vr_buf   = (float*) carve((size_t)BS * NL * 2 * sizeof(float));

  auto cdiv = [](int a, int b) { return (a + b - 1) / b; };
  const int TB = 256;
  const int HWs[4]   = {10000, 2500, 625, 169};
  const int basel[4] = {0, 10000, 12500, 13125};

  // ---- prep ----
  vr_kernel<<<1, 32, 0, stream>>>(masks[0], masks[1], masks[2], masks[3], vr_buf);
  for (int l = 0; l < 4; ++l) {
    int n = BS * HWs[l] * EMBED;
    flatten_kernel<<<cdiv(n, TB), TB, 0, stream>>>(feats[l], featflat, HWs[l], basel[l]);
  }
  ref_kernel<<<cdiv(MQ * NL, TB), TB, 0, stream>>>(ref_points, vr_buf, ref_buf);
  qpos_kernel<<<cdiv(MQ * EMBED, TB), TB, 0, stream>>>(ref_buf, qpos);
  init_query_kernel<<<cdiv(MQ * EMBED, TB), TB, 0, stream>>>(cls_embed, query, MQ * EMBED);

  const int NE = MQ * EMBED;              // 2097152
  for (int l = 0; l < NLAYERS; ++l) {
    // value projection: v = feat_flat @ vp_w[l] + vp_b[l]   (masks all-false -> no zeroing)
    wconv_t_kernel<<<cdiv(EMBED * EMBED, TB), TB, 0, stream>>>(vp_w + l * EMBED * EMBED, wt_bf, EMBED);
    gemm_bf16_wmma<<<dim3(EMBED / 64, cdiv(MV, 128)), TB, 0, stream>>>(
        featflat, wt_bf, vp_b + l * EMBED, v_buf, MV, EMBED);

    // q = query + qpos (bf16 operand)
    add_to_bf16_kernel<<<cdiv(NE, TB), TB, 0, stream>>>(query, qpos, q_bf, NE);

    // sampling offsets
    wconv_t_kernel<<<cdiv(EMBED * EMBED, TB), TB, 0, stream>>>(so_w + l * EMBED * EMBED, wt_bf, EMBED);
    gemm_bf16_wmma<<<dim3(EMBED / 64, MQ / 128), TB, 0, stream>>>(
        q_bf, wt_bf, so_b + l * EMBED, so_buf, MQ, EMBED);

    // attention weights (N = 128) + softmax over 16
    wconv_t_kernel<<<cdiv(128 * EMBED, TB), TB, 0, stream>>>(aw_w + l * EMBED * 128, wt_bf, 128);
    gemm_bf16_wmma<<<dim3(128 / 64, MQ / 128), TB, 0, stream>>>(
        q_bf, wt_bf, aw_b + l * 128, aw_buf, MQ, 128);
    softmax16_kernel<<<cdiv(MQ * NH, TB), TB, 0, stream>>>(aw_buf);

    // deformable sampling
    msda_kernel<<<cdiv(MQ * NH * 32, TB), TB, 0, stream>>>(v_buf, ref_buf, so_buf, aw_buf, msda_bf);

    // output projection + residual
    wconv_t_kernel<<<cdiv(EMBED * EMBED, TB), TB, 0, stream>>>(op_w + l * EMBED * EMBED, wt_bf, EMBED);
    gemm_bf16_wmma<<<dim3(EMBED / 64, MQ / 128), TB, 0, stream>>>(
        msda_bf, wt_bf, op_b + l * EMBED, op_out, MQ, EMBED);
    residual_kernel<<<cdiv(NE, TB), TB, 0, stream>>>(op_out, query, qn_bf, NE);

    // FFN (no residual); last layer writes d_out directly
    wconv_t_kernel<<<cdiv(EMBED * EMBED, TB), TB, 0, stream>>>(ffn_w + l * EMBED * EMBED, wt_bf, EMBED);
    float* ffn_dst = (l == NLAYERS - 1) ? (float*)d_out : query;
    gemm_bf16_wmma<<<dim3(EMBED / 64, MQ / 128), TB, 0, stream>>>(
        qn_bf, wt_bf, ffn_b + l * EMBED, ffn_dst, MQ, EMBED);
  }
}
